// UNetV2_44547400794350
// MI455X (gfx1250) — matrix operand order repaired and checked by reading, not compiled, not run on hardware
//
#include <hip/hip_runtime.h>
#include <hip/hip_bf16.h>

// ---------------------------------------------------------------------------
// CDNA5 (gfx1250) dense-emulated sparse UNet.
// Every conv (stride-2 "down", input-dilated "inverse", (3,1,1) head) is an
// implicit GEMM on v_wmma_f32_16x16x32_f16:
//   - wave32 computes 16 voxels x (16*NPW) couts, NPW in {1,2,4}
//   - <NPW, KW3, INV, C4> are template params -> zero dynamic branches in the
//     hot loop; all index math is shift/mask + magic mul/shift tap decode
//   - A gather is branchless: validity is data (cndmask), addresses clamped
//     to a safe in-bounds location
//   - B fragments: weights pre-swizzled into per-lane WMMA order ->
//     one contiguous 32B load per lane per chunk
//   - 2-deep software pipeline with an EVEN padded chunk count and zero-padded
//     B panels: single-exit straight-line loop body, so accumulators stay in
//     one register tuple (no v_mov walls) and s_wait_loadcnt only covers the
//     previous iteration's loads
//   - epilogue fuses BN affine + optional residual + ReLU + sparse mask
// ---------------------------------------------------------------------------

typedef __attribute__((ext_vector_type(16))) _Float16 v16h;
typedef __attribute__((ext_vector_type(8)))  float    v8f;
typedef __attribute__((ext_vector_type(4)))  float    v4f;

#define EPSBN 1e-3f
#define NB 2  // batch

struct ConvArgs {
  const float* in1;        // first input tensor (cin1 channels)
  const float* in2;        // optional channel-concat second input (cin2 ch)
  const _Float16* wqs;     // swizzled f16 weight panels
  const float* s;          // BN scale per cout
  const float* t;          // BN shift per cout
  const float* resid;      // optional residual added before ReLU
  const float* mask;       // optional active-site mask per out voxel
  float* out;
  int cin1, cin2;
  int ciShift, ciMask;     // Ci = 1<<ciShift
  int Di, Hi, Wi;
  int Do, Ho, Wo, Co;
  int sd, sh, sw;
  int pd, ph, pw;          // pad-lo
  int KT;                  // taps*Ci
  int chunksPad;           // even loop bound
  int chunksA;             // allocated panels per n-tile (chunksPad+2)
};

template <bool KW3>
__device__ __forceinline__ void tap_decode(int tap, int& td, int& th, int& tw) {
  if (KW3) {                 // taps < 32: exact magic-constant div by 9 and 3
    td = (tap * 57) >> 9;
    int t2 = tap - td * 9;
    th = (t2 * 11) >> 5;
    tw = t2 - th * 3;
  } else {                   // (3,1,1) head
    td = tap; th = 0; tw = 0;
  }
}

// Branchless: returns clamped-safe voxel index, validity as data in `ok`.
template <bool INV>
__device__ __forceinline__ int tap_addr(const ConvArgs& a, int bz, int oz,
                                        int oy, int ox, int td, int th, int tw,
                                        bool& ok) {
  int zd = oz * a.sd + td - a.pd;
  int yd = oy * a.sh + th - a.ph;
  int xd = ox * a.sw + tw - a.pw;
  bool v = true;
  if (INV) {                           // input dilation 2 in all three dims
    v = !((zd | yd | xd) & 1);
    zd >>= 1; yd >>= 1; xd >>= 1;
  }
  v = v && ((unsigned)zd < (unsigned)a.Di) &&
      ((unsigned)yd < (unsigned)a.Hi) && ((unsigned)xd < (unsigned)a.Wi);
  int vin = ((bz * a.Di + zd) * a.Hi + yd) * a.Wi + xd;
  ok = v;
  return v ? vin : 0;  // voxel 0 is always a safe address
}

// Ci >= 16: 8-K segment = 8 contiguous channels of one tap -> 2x b128,
// unconditional load from clamped address, zeroed by cndmask.
template <bool KW3, bool INV>
__device__ __forceinline__ v16h loadA_fast(const ConvArgs& a, int kb,
                                           int laneHi, bool rowValid, int bz,
                                           int oz, int oy, int ox) {
  v16h A;
#pragma unroll
  for (int seg = 0; seg < 2; ++seg) {
    int kStart = kb + (laneHi << 3) + (seg << 4);
    int tap = kStart >> a.ciShift;
    int c0  = kStart & a.ciMask;
    int td, th, tw;
    tap_decode<KW3>(tap, td, th, tw);
    bool tok;
    int vin = tap_addr<INV>(a, bz, oz, oy, ox, td, th, tw, tok);
    bool ok = rowValid && (kStart < a.KT) && tok;
    if (!ok) vin = 0;
    const float* p = (c0 < a.cin1)
                         ? a.in1 + (size_t)vin * a.cin1 + c0
                         : a.in2 + (size_t)vin * a.cin2 + (c0 - a.cin1);
    v4f u0 = ((const v4f*)p)[0];
    v4f u1 = ((const v4f*)p)[1];
#pragma unroll
    for (int e = 0; e < 4; ++e) {
      A[(seg << 3) + e]     = (_Float16)(ok ? u0[e] : 0.0f);
      A[(seg << 3) + 4 + e] = (_Float16)(ok ? u1[e] : 0.0f);
    }
  }
  return A;
}

// Ci == 4 (conv_input only): 8-K segment = 2 taps x 4 channels, branchless
__device__ __forceinline__ v16h loadA_c4(const ConvArgs& a, int kb, int laneHi,
                                         bool rowValid, int bz, int oz, int oy,
                                         int ox) {
  v16h A;
#pragma unroll
  for (int seg = 0; seg < 2; ++seg) {
    int kStart = kb + (laneHi << 3) + (seg << 4);
#pragma unroll
    for (int half = 0; half < 2; ++half) {
      int k0 = kStart + (half << 2);
      int tap = k0 >> 2;
      int td, th, tw;
      tap_decode<true>(tap, td, th, tw);
      bool tok;
      int vin = tap_addr<false>(a, bz, oz, oy, ox, td, th, tw, tok);
      bool ok = rowValid && (k0 < a.KT) && tok;
      if (!ok) vin = 0;
      v4f u = *(const v4f*)(a.in1 + ((size_t)vin << 2));
#pragma unroll
      for (int e = 0; e < 4; ++e)
        A[(seg << 3) + (half << 2) + e] = (_Float16)(ok ? u[e] : 0.0f);
    }
  }
  return A;
}

template <int NPW, bool KW3, bool INV, bool C4>
__global__ __launch_bounds__(256)
void conv3d_wmma_k(ConvArgs a, int totalWaves, int ngMask, int ngShift) {
  const int wg = blockIdx.x * (blockDim.x >> 5) + ((int)threadIdx.x >> 5);
  if (wg >= totalWaves) return;            // wave-uniform: EXEC stays full
  const int lane   = threadIdx.x & 31;
  const int laneHi = lane >> 4;
  const int lanelo = lane & 15;
  const int ng = wg & ngMask;
  const int mt = wg >> ngShift;

  const int totalVox = NB * a.Do * a.Ho * a.Wo;
  const int vA = mt * 16 + lanelo;
  const bool rowValid = vA < totalVox;
  int bz = 0, oz = 0, oy = 0, ox = 0;
  {
    int r = rowValid ? vA : 0;
    ox = r % a.Wo; r /= a.Wo;
    oy = r % a.Ho; r /= a.Ho;
    oz = r % a.Do; r /= a.Do;
    bz = r;
  }

  v8f acc[NPW] = {};
  const _Float16* bbase =
      a.wqs + (((size_t)ng * NPW * a.chunksA) << 9) + (lane << 4);

#define LDB(ch, j) \
  (*(const v16h*)(bbase + (((size_t)((j) * a.chunksA + (ch))) << 9)))
#define LDA(ch)                                                          \
  (C4 ? loadA_c4(a, (ch) << 5, laneHi, rowValid, bz, oz, oy, ox)         \
      : loadA_fast<KW3, INV>(a, (ch) << 5, laneHi, rowValid, bz, oz, oy, ox))

  // 2-deep pipeline; chunksPad even, panels zero-padded -> single-exit loop,
  // unconditional prefetch, no accumulator shuffling.
  v16h A0, A1;
  v16h B0[NPW], B1[NPW];
  A0 = LDA(0);
#pragma unroll
  for (int j = 0; j < NPW; ++j) B0[j] = LDB(0, j);

  for (int ch = 0; ch < a.chunksPad; ch += 2) {
    A1 = LDA(ch + 1);
#pragma unroll
    for (int j = 0; j < NPW; ++j) B1[j] = LDB(ch + 1, j);
#pragma unroll
    for (int j = 0; j < NPW; ++j)
      acc[j] = __builtin_amdgcn_wmma_f32_16x16x32_f16(
          false, A0, false, B0[j], (short)0, acc[j], false, false);
    A0 = LDA(ch + 2);
#pragma unroll
    for (int j = 0; j < NPW; ++j) B0[j] = LDB(ch + 2, j);
#pragma unroll
    for (int j = 0; j < NPW; ++j)
      acc[j] = __builtin_amdgcn_wmma_f32_16x16x32_f16(
          false, A1, false, B1[j], (short)0, acc[j], false, false);
  }
#undef LDA
#undef LDB

  // epilogue: BN affine + residual + ReLU + sparse mask
#pragma unroll
  for (int j = 0; j < NPW; ++j) {
    const int nBase = ((ng * NPW + j) << 4) + lanelo;
    const float sc = a.s[nBase];
    const float sh = a.t[nBase];
#pragma unroll
    for (int r = 0; r < 8; ++r) {
      int v = mt * 16 + r + (laneHi << 3);
      if (v < totalVox) {
        float o = acc[j][r] * sc + sh;
        size_t off = (size_t)v * a.Co + nBase;
        if (a.resid) o += a.resid[off];
        o = fmaxf(o, 0.0f);
        if (a.mask) o *= a.mask[v];
        a.out[off] = o;
      }
    }
  }
}

// ---------------------------------------------------------------------------
// Support kernels
// ---------------------------------------------------------------------------

__global__ void fill0_k(float* p, int n) {
  int i = blockIdx.x * blockDim.x + threadIdx.x;
  if (i < n) p[i] = 0.0f;
}

__global__ void scatter_k(const float* __restrict__ feats,
                          const int* __restrict__ coords,
                          float* __restrict__ x0, float* __restrict__ mask,
                          int n, int D, int H, int W, int C) {
  int i = blockIdx.x * blockDim.x + threadIdx.x;
  if (i >= n) return;
  int b = coords[4 * i + 0];
  int z = coords[4 * i + 1];
  int y = coords[4 * i + 2];
  int x = coords[4 * i + 3];
  int v = ((b * D + z) * H + y) * W + x;
  for (int c = 0; c < C; ++c)
    atomicAdd(&x0[(size_t)v * C + c], feats[(size_t)i * C + c]);
  mask[v] = 1.0f;
}

struct MaskArgs {
  const float* mi; float* mo;
  int Di, Hi, Wi, Do, Ho, Wo;
  int kd, kh, kw, sd, sh, sw, pd, ph, pw;
};

__global__ void mask_down_k(MaskArgs m) {
  int i = blockIdx.x * blockDim.x + threadIdx.x;
  int total = NB * m.Do * m.Ho * m.Wo;
  if (i >= total) return;
  int r = i;
  int x = r % m.Wo; r /= m.Wo;
  int y = r % m.Ho; r /= m.Ho;
  int z = r % m.Do; r /= m.Do;
  int b = r;
  float on = 0.0f;
  for (int td = 0; td < m.kd; ++td) {
    int zi = z * m.sd + td - m.pd;
    if ((unsigned)zi >= (unsigned)m.Di) continue;
    for (int th = 0; th < m.kh; ++th) {
      int yi = y * m.sh + th - m.ph;
      if ((unsigned)yi >= (unsigned)m.Hi) continue;
      for (int tw = 0; tw < m.kw; ++tw) {
        int xi = x * m.sw + tw - m.pw;
        if ((unsigned)xi >= (unsigned)m.Wi) continue;
        if (m.mi[((b * m.Di + zi) * m.Hi + yi) * m.Wi + xi] > 0.0f) on = 1.0f;
      }
    }
  }
  m.mo[i] = on;
}

// h[v,j] = xm[v,j] + channel_reduction(concat(bot, top))[v,j]
__global__ void add_chred_k(const float* __restrict__ xm,
                            const float* __restrict__ bot,
                            const float* __restrict__ top,
                            float* __restrict__ h, int total, int C) {
  int i = blockIdx.x * blockDim.x + threadIdx.x;
  if (i >= total * C) return;
  int v = i / C, j = i % C;
  int half = C >> 1;
  float red;
  if (j < half) {
    red = bot[(size_t)v * C + 2 * j] + bot[(size_t)v * C + 2 * j + 1];
  } else {
    int jj = 2 * (j - half);
    red = top[(size_t)v * C + jj] + top[(size_t)v * C + jj + 1];
  }
  h[i] = xm[i] + red;
}

// Fold BN into (s,t) and write f16 weights in swizzled WMMA-B panel order:
// block (nt, ch) of 512 f16 = [lane 0..31][e 0..15];
// K = ch*32 + (lane>>4)*8 + (e>>3)*16 + (e&7), col = nt*16 + (lane&15); K >= KT -> 0 pad
__global__ void prep_swz_k(const float* __restrict__ w,
                           const float* __restrict__ g,
                           const float* __restrict__ b,
                           const float* __restrict__ m,
                           const float* __restrict__ v,
                           _Float16* __restrict__ wqs, float* __restrict__ s,
                           float* __restrict__ t, int KT, int Co, int chunksA) {
  int i = blockIdx.x * blockDim.x + threadIdx.x;
  if (i < Co) {
    float sc = g[i] * rsqrtf(v[i] + EPSBN);
    s[i] = sc;
    t[i] = b[i] - m[i] * sc;
  }
  int total = (Co >> 4) * chunksA * 512;
  if (i >= total) return;
  int e    = i & 15;
  int lane = (i >> 4) & 31;
  int blk  = i >> 9;
  int ch   = blk % chunksA;
  int nt   = blk / chunksA;
  int K    = (ch << 5) + ((lane >> 4) << 3) + ((e >> 3) << 4) + (e & 7);
  int col  = (nt << 4) + (lane & 15);
  wqs[i] = (K < KT) ? (_Float16)w[(size_t)K * Co + col] : (_Float16)0.0f;
}

// ---------------------------------------------------------------------------
// Host orchestration
// ---------------------------------------------------------------------------

static inline int ilog2(int x) { int l = 0; while ((1 << l) < x) ++l; return l; }
static inline int chunks_alloc(int KT) {
  int cr = (KT + 31) >> 5;
  int cp = (cr + 1) & ~1;
  return cp + 2;
}

static inline void conv_go(hipStream_t st, const float* in1, const float* in2,
                           int cin1, int Ci, const _Float16* wqs,
                           const float* s, const float* t, const float* resid,
                           const float* mask, float* out, int Di, int Hi,
                           int Wi, int Do, int Ho, int Wo, int Co, int taps,
                           bool kw3, int sd, int sh, int sw, int pd, int ph,
                           int pw, bool inv) {
  ConvArgs a;
  a.in1 = in1; a.in2 = in2; a.wqs = wqs; a.s = s; a.t = t;
  a.resid = resid; a.mask = mask; a.out = out;
  a.cin1 = cin1; a.cin2 = Ci - cin1;
  a.ciShift = ilog2(Ci); a.ciMask = Ci - 1;
  a.Di = Di; a.Hi = Hi; a.Wi = Wi;
  a.Do = Do; a.Ho = Ho; a.Wo = Wo; a.Co = Co;
  a.sd = sd; a.sh = sh; a.sw = sw;
  a.pd = pd; a.ph = ph; a.pw = pw;
  a.KT = taps * Ci;
  {
    int cr = (a.KT + 31) >> 5;
    a.chunksPad = (cr + 1) & ~1;
    a.chunksA = a.chunksPad + 2;
  }

  int totalVox = NB * Do * Ho * Wo;
  int mTiles = (totalVox + 15) >> 4;
  int nT = Co >> 4;
  int NPW = nT >= 4 ? 4 : nT;
  int nGroups = nT / NPW;
  int ngShift = ilog2(nGroups);
  int totalWaves = mTiles * nGroups;
  int blocks = (totalWaves + 7) / 8;
#define LAUNCH(N, K3, IV, C4)                             \
  conv3d_wmma_k<N, K3, IV, C4>                            \
      <<<blocks, 256, 0, st>>>(a, totalWaves, nGroups - 1, ngShift)
  if (Ci == 4)       LAUNCH(1, true, false, true);
  else if (!kw3)     LAUNCH(4, false, false, false);   // conv_out, Co=128
  else if (inv) {
    if (NPW == 1)      LAUNCH(1, true, true, false);
    else if (NPW == 2) LAUNCH(2, true, true, false);
    else               LAUNCH(4, true, true, false);
  } else {
    if (NPW == 1)      LAUNCH(1, true, false, false);
    else if (NPW == 2) LAUNCH(2, true, false, false);
    else               LAUNCH(4, true, false, false);
  }
#undef LAUNCH
}

static inline void mask_go(hipStream_t st, const float* mi, float* mo, int Di,
                           int Hi, int Wi, int Do, int Ho, int Wo, int kd,
                           int kh, int kw, int sd, int sh, int sw, int pd,
                           int ph, int pw) {
  MaskArgs m;
  m.mi = mi; m.mo = mo;
  m.Di = Di; m.Hi = Hi; m.Wi = Wi; m.Do = Do; m.Ho = Ho; m.Wo = Wo;
  m.kd = kd; m.kh = kh; m.kw = kw; m.sd = sd; m.sh = sh; m.sw = sw;
  m.pd = pd; m.ph = ph; m.pw = pw;
  int total = NB * Do * Ho * Wo;
  mask_down_k<<<(total + 255) / 256, 256, 0, st>>>(m);
}

extern "C" void kernel_launch(void* const* d_in, const int* in_sizes, int n_in,
                              void* d_out, int out_size, void* d_ws,
                              size_t ws_size, hipStream_t stream) {
  (void)n_in; (void)out_size; (void)ws_size;
  const int D1 = 25, H1 = 96, W1 = 96; const int V1 = NB * D1 * H1 * W1;
  const int D2 = 13, H2 = 48, W2 = 48; const int V2 = NB * D2 * H2 * W2;
  const int D3 = 7,  H3 = 24, W3 = 24; const int V3 = NB * D3 * H3 * W3;
  const int D4 = 3,  H4 = 12, W4 = 12; const int V4 = NB * D4 * H4 * W4;
  const int DO = 1,  HO = 12, WO = 12; const int VO = NB * DO * HO * WO;

  const float* feats  = (const float*)d_in[0];
  const int*   coords = (const int*)d_in[1];
  const int    nvox   = in_sizes[0] / 4;

  struct LP { const float *w, *g, *b, *m, *v; };
  LP lp[28];
  {
    int idx = 2;
    for (int i = 0; i < 28; ++i) {
      lp[i].w = (const float*)d_in[idx + 0];
      lp[i].g = (const float*)d_in[idx + 1];
      lp[i].b = (const float*)d_in[idx + 2];
      lp[i].m = (const float*)d_in[idx + 3];
      lp[i].v = (const float*)d_in[idx + 4];
      idx += 5;
    }
  }
  // {taps, cin, cout} in params insertion order
  static const int LT[28][3] = {
      {27, 4, 16},   {27, 16, 16},  {27, 16, 32},  {27, 32, 32},  {27, 32, 32},
      {27, 32, 64},  {27, 64, 64},  {27, 64, 64},  {27, 64, 64},  {27, 64, 64},
      {27, 64, 64},  {27, 128, 64}, {27, 64, 64},  {27, 128, 64}, {27, 64, 32},
      {27, 64, 32},  {27, 32, 16},  {27, 32, 16},  {27, 16, 16},  {3, 64, 128},
      {27, 64, 64},  {27, 64, 64},  {27, 64, 64},  {27, 64, 64},  {27, 32, 32},
      {27, 32, 32},  {27, 16, 16},  {27, 16, 16},
  };

  char* wsp = (char*)d_ws;
  auto alloc = [&](size_t bytes) -> void* {
    void* p = (void*)wsp;
    wsp += (bytes + 255) & ~(size_t)255;
    return p;
  };

  // per-layer swizzled f16 weight panels + folded BN
  _Float16* wq[28]; float* ls[28]; float* lt[28];
  for (int i = 0; i < 28; ++i) {
    int KT = LT[i][0] * LT[i][1];
    int co = LT[i][2];
    int chunksA = chunks_alloc(KT);
    int total = (co >> 4) * chunksA * 512;
    wq[i] = (_Float16*)alloc((size_t)total * sizeof(_Float16));
    ls[i] = (float*)alloc((size_t)co * sizeof(float));
    lt[i] = (float*)alloc((size_t)co * sizeof(float));
    int n = total > co ? total : co;
    prep_swz_k<<<(n + 255) / 256, 256, 0, stream>>>(
        lp[i].w, lp[i].g, lp[i].b, lp[i].m, lp[i].v, wq[i], ls[i], lt[i], KT,
        co, chunksA);
  }

  // activations (fp32, NDHWC)
  float* x0    = (float*)alloc((size_t)V1 * 4 * 4);
  float* mask1 = (float*)alloc((size_t)V1 * 4);
  float* a1    = (float*)alloc((size_t)V1 * 16 * 4);  // reused: conv_input, e1, g1
  float* x1    = (float*)alloc((size_t)V1 * 16 * 4);
  float* u2    = (float*)alloc((size_t)V1 * 16 * 4);
  float* f1    = (float*)alloc((size_t)V1 * 16 * 4);
  float* h1    = (float*)alloc((size_t)V1 * 16 * 4);
  float* mask2 = (float*)alloc((size_t)V2 * 4);
  float* tA2   = (float*)alloc((size_t)V2 * 32 * 4);
  float* tB2   = (float*)alloc((size_t)V2 * 32 * 4);
  float* x2    = (float*)alloc((size_t)V2 * 32 * 4);
  float* u3    = (float*)alloc((size_t)V2 * 32 * 4);
  float* h2    = (float*)alloc((size_t)V2 * 32 * 4);
  float* mask3 = (float*)alloc((size_t)V3 * 4);
  float* tA3   = (float*)alloc((size_t)V3 * 64 * 4);
  float* tB3   = (float*)alloc((size_t)V3 * 64 * 4);
  float* x3    = (float*)alloc((size_t)V3 * 64 * 4);
  float* u4    = (float*)alloc((size_t)V3 * 64 * 4);
  float* h3    = (float*)alloc((size_t)V3 * 64 * 4);
  float* mask4 = (float*)alloc((size_t)V4 * 4);
  float* tA4   = (float*)alloc((size_t)V4 * 64 * 4);
  float* tB4   = (float*)alloc((size_t)V4 * 64 * 4);
  float* x4    = (float*)alloc((size_t)V4 * 64 * 4);
  float* h4    = (float*)alloc((size_t)V4 * 64 * 4);
  float* maskO = (float*)alloc((size_t)VO * 4);

  float* out_head = (float*)d_out;                     // (2,1,12,12,128)
  float* out_u1   = (float*)d_out + (size_t)VO * 128;  // (2,25,96,96,16)

  // scatter voxels
  fill0_k<<<(V1 * 4 + 255) / 256, 256, 0, stream>>>(x0, V1 * 4);
  fill0_k<<<(V1 + 255) / 256, 256, 0, stream>>>(mask1, V1);
  scatter_k<<<(nvox + 255) / 256, 256, 0, stream>>>(feats, coords, x0, mask1,
                                                    nvox, D1, H1, W1, 4);

  // encoder
  conv_go(stream, x0, nullptr, 4, 4, wq[0], ls[0], lt[0], nullptr, mask1, a1,
          D1, H1, W1, D1, H1, W1, 16, 27, true, 1, 1, 1, 1, 1, 1, false);
  conv_go(stream, a1, nullptr, 16, 16, wq[1], ls[1], lt[1], nullptr, mask1, x1,
          D1, H1, W1, D1, H1, W1, 16, 27, true, 1, 1, 1, 1, 1, 1, false);

  mask_go(stream, mask1, mask2, D1, H1, W1, D2, H2, W2, 3, 3, 3, 2, 2, 2, 1, 1, 1);
  conv_go(stream, x1, nullptr, 16, 16, wq[2], ls[2], lt[2], nullptr, mask2, tA2,
          D1, H1, W1, D2, H2, W2, 32, 27, true, 2, 2, 2, 1, 1, 1, false);
  conv_go(stream, tA2, nullptr, 32, 32, wq[3], ls[3], lt[3], nullptr, mask2, tB2,
          D2, H2, W2, D2, H2, W2, 32, 27, true, 1, 1, 1, 1, 1, 1, false);
  conv_go(stream, tB2, nullptr, 32, 32, wq[4], ls[4], lt[4], nullptr, mask2, x2,
          D2, H2, W2, D2, H2, W2, 32, 27, true, 1, 1, 1, 1, 1, 1, false);

  mask_go(stream, mask2, mask3, D2, H2, W2, D3, H3, W3, 3, 3, 3, 2, 2, 2, 1, 1, 1);
  conv_go(stream, x2, nullptr, 32, 32, wq[5], ls[5], lt[5], nullptr, mask3, tA3,
          D2, H2, W2, D3, H3, W3, 64, 27, true, 2, 2, 2, 1, 1, 1, false);
  conv_go(stream, tA3, nullptr, 64, 64, wq[6], ls[6], lt[6], nullptr, mask3, tB3,
          D3, H3, W3, D3, H3, W3, 64, 27, true, 1, 1, 1, 1, 1, 1, false);
  conv_go(stream, tB3, nullptr, 64, 64, wq[7], ls[7], lt[7], nullptr, mask3, x3,
          D3, H3, W3, D3, H3, W3, 64, 27, true, 1, 1, 1, 1, 1, 1, false);

  mask_go(stream, mask3, mask4, D3, H3, W3, D4, H4, W4, 3, 3, 3, 2, 2, 2, 0, 1, 1);
  conv_go(stream, x3, nullptr, 64, 64, wq[8], ls[8], lt[8], nullptr, mask4, tA4,
          D3, H3, W3, D4, H4, W4, 64, 27, true, 2, 2, 2, 0, 1, 1, false);
  conv_go(stream, tA4, nullptr, 64, 64, wq[9], ls[9], lt[9], nullptr, mask4, tB4,
          D4, H4, W4, D4, H4, W4, 64, 27, true, 1, 1, 1, 1, 1, 1, false);
  conv_go(stream, tB4, nullptr, 64, 64, wq[10], ls[10], lt[10], nullptr, mask4, x4,
          D4, H4, W4, D4, H4, W4, 64, 27, true, 1, 1, 1, 1, 1, 1, false);

  // output head conv_out: (3,1,1), stride (2,1,1), pad 0
  mask_go(stream, mask4, maskO, D4, H4, W4, DO, HO, WO, 3, 1, 1, 2, 1, 1, 0, 0, 0);
  conv_go(stream, x4, nullptr, 64, 64, wq[19], ls[19], lt[19], nullptr, maskO,
          out_head, D4, H4, W4, DO, HO, WO, 128, 3, false, 2, 1, 1, 0, 0, 0, false);

  // decoder: level 4
  conv_go(stream, x4, nullptr, 64, 64, wq[20], ls[20], lt[20], nullptr, mask4, tA4,
          D4, H4, W4, D4, H4, W4, 64, 27, true, 1, 1, 1, 1, 1, 1, false);
  conv_go(stream, tA4, nullptr, 64, 64, wq[21], ls[21], lt[21], x4, mask4, tB4,
          D4, H4, W4, D4, H4, W4, 64, 27, true, 1, 1, 1, 1, 1, 1, false);
  conv_go(stream, x4, tB4, 64, 128, wq[11], ls[11], lt[11], nullptr, mask4, tA4,
          D4, H4, W4, D4, H4, W4, 64, 27, true, 1, 1, 1, 1, 1, 1, false);
  add_chred_k<<<(V4 * 64 + 255) / 256, 256, 0, stream>>>(tA4, x4, tB4, h4, V4, 64);
  conv_go(stream, h4, nullptr, 64, 64, wq[12], ls[12], lt[12], nullptr, mask3, u4,
          D4, H4, W4, D3, H3, W3, 64, 27, true, 1, 1, 1, 2, 1, 1, true);

  // level 3
  conv_go(stream, x3, nullptr, 64, 64, wq[22], ls[22], lt[22], nullptr, mask3, tA3,
          D3, H3, W3, D3, H3, W3, 64, 27, true, 1, 1, 1, 1, 1, 1, false);
  conv_go(stream, tA3, nullptr, 64, 64, wq[23], ls[23], lt[23], x3, mask3, tB3,
          D3, H3, W3, D3, H3, W3, 64, 27, true, 1, 1, 1, 1, 1, 1, false);
  conv_go(stream, u4, tB3, 64, 128, wq[13], ls[13], lt[13], nullptr, mask3, tA3,
          D3, H3, W3, D3, H3, W3, 64, 27, true, 1, 1, 1, 1, 1, 1, false);
  add_chred_k<<<(V3 * 64 + 255) / 256, 256, 0, stream>>>(tA3, u4, tB3, h3, V3, 64);
  conv_go(stream, h3, nullptr, 64, 64, wq[14], ls[14], lt[14], nullptr, mask2, u3,
          D3, H3, W3, D2, H2, W2, 32, 27, true, 1, 1, 1, 1, 1, 1, true);

  // level 2
  conv_go(stream, x2, nullptr, 32, 32, wq[24], ls[24], lt[24], nullptr, mask2, tA2,
          D2, H2, W2, D2, H2, W2, 32, 27, true, 1, 1, 1, 1, 1, 1, false);
  conv_go(stream, tA2, nullptr, 32, 32, wq[25], ls[25], lt[25], x2, mask2, tB2,
          D2, H2, W2, D2, H2, W2, 32, 27, true, 1, 1, 1, 1, 1, 1, false);
  conv_go(stream, u3, tB2, 32, 64, wq[15], ls[15], lt[15], nullptr, mask2, tA2,
          D2, H2, W2, D2, H2, W2, 32, 27, true, 1, 1, 1, 1, 1, 1, false);
  add_chred_k<<<(V2 * 32 + 255) / 256, 256, 0, stream>>>(tA2, u3, tB2, h2, V2, 32);
  conv_go(stream, h2, nullptr, 32, 32, wq[16], ls[16], lt[16], nullptr, mask1, u2,
          D2, H2, W2, D1, H1, W1, 16, 27, true, 1, 1, 1, 1, 1, 1, true);

  // level 1
  conv_go(stream, x1, nullptr, 16, 16, wq[26], ls[26], lt[26], nullptr, mask1, a1,
          D1, H1, W1, D1, H1, W1, 16, 27, true, 1, 1, 1, 1, 1, 1, false);
  conv_go(stream, a1, nullptr, 16, 16, wq[27], ls[27], lt[27], x1, mask1, f1,
          D1, H1, W1, D1, H1, W1, 16, 27, true, 1, 1, 1, 1, 1, 1, false);
  conv_go(stream, u2, f1, 16, 32, wq[17], ls[17], lt[17], nullptr, mask1, a1,
          D1, H1, W1, D1, H1, W1, 16, 27, true, 1, 1, 1, 1, 1, 1, false);
  add_chred_k<<<(V1 * 16 + 255) / 256, 256, 0, stream>>>(a1, u2, f1, h1, V1, 16);
  conv_go(stream, h1, nullptr, 16, 16, wq[18], ls[18], lt[18], nullptr, mask1,
          out_u1, D1, H1, W1, D1, H1, W1, 16, 27, true, 1, 1, 1, 1, 1, 1, false);
}
